// Conv7by3DirectConv3Padding2_51110110822427
// MI455X (gfx1250) — 8-bit matrix operand order repaired and checked by reading, compiled, not run on hardware
//
#include <hip/hip_runtime.h>
#include <stdint.h>

typedef __attribute__((ext_vector_type(8))) int v8i;

#define N_IMG 16
#define C_IN  64
#define H_IN  112
#define W_IN  112
#define K_OUT 128
#define H_OUT 106
#define W_OUT 106
#define TAPS  49

#define LDS_ROW    4480
#define LDS_BYTES  (7 * LDS_ROW)
#define LDS_CHUNKS (LDS_BYTES / 16)

__global__ __launch_bounds__(128) void pack_x_kernel(const int* __restrict__ x,
                                                     unsigned char* __restrict__ xu8) {
  const int nh = blockIdx.x;
  const int n = nh / H_IN, h = nh % H_IN;
  const int w = threadIdx.x;
  if (w >= W_IN) return;
  unsigned int dw[16];
#pragma unroll
  for (int d = 0; d < 16; ++d) dw[d] = 0u;
#pragma unroll
  for (int c = 0; c < C_IN; ++c) {
    int v = x[(((size_t)(n * C_IN + c) * H_IN + h) * W_IN) + w];
    v = v < 0 ? 0 : (v > 7 ? 7 : v);
    dw[c >> 2] |= ((unsigned int)v) << ((c & 3) * 8);
  }
  uint4* dst = (uint4*)(xu8 + (((size_t)(n * H_IN + h)) * W_IN + w) * C_IN);
#pragma unroll
  for (int q = 0; q < 4; ++q)
    dst[q] = make_uint4(dw[4 * q + 0], dw[4 * q + 1], dw[4 * q + 2], dw[4 * q + 3]);
}

__device__ __forceinline__ float wtap(const float* w1, const float* w2,
                                      const float* w3, int k, int c, int r, int s) {
  const int base = ((k * C_IN + c) * 7 + r) * 3;
  if (s < 3)  return w1[base + s];
  if (s == 3) return w2[base + 1];
  return w3[base + (s - 4)];
}

__global__ __launch_bounds__(256) void pack_w_kernel(const float* __restrict__ w1,
                                                     const float* __restrict__ w2,
                                                     const float* __restrict__ w3,
                                                     unsigned int* __restrict__ wpack) {
  const int t = blockIdx.x;
  const int r = t / 7, s = t % 7;
  const int b = threadIdx.x >> 5;
  const int lane = threadIdx.x & 31;
  const int k = b * 16 + (lane & 15);
  const int khalf = lane >> 4;
  unsigned int* dst = wpack + ((size_t)((t * 8 + b) * 32 + lane) * 8);
#pragma unroll
  for (int v = 0; v < 8; ++v) {
    const int c0 = (v >> 1) * 16 + khalf * 8 + (v & 1) * 4;
    unsigned int dwv = 0;
#pragma unroll
    for (int bb = 0; bb < 4; ++bb) {
      const unsigned int q = (unsigned int)wtap(w1, w2, w3, k, c0 + bb, r, s);
      dwv |= (q & 0xFFu) << (bb * 8);
    }
    dst[v] = dwv;
  }
}

__global__ __launch_bounds__(256) void conv7x7_iu8_wmma(
    const unsigned char* __restrict__ xu8, const uint4* __restrict__ wpack,
    float* __restrict__ out) {
  __shared__ __align__(16) unsigned char smem[LDS_BYTES];

  const int n  = blockIdx.z;
  const int io = blockIdx.y;
  const int j0 = (blockIdx.x == 0) ? 0 : (W_OUT - 64);
  const int tid = threadIdx.x;

  {
    const unsigned char* src =
        xu8 + (((size_t)(n * H_IN + io)) * W_IN + j0) * C_IN;
#pragma unroll
    for (int it = 0; it < 8; ++it) {
      const int q = it * 256 + tid;
      if (q < LDS_CHUNKS) {
        const int rowi = q / 280;
        const int off16 = (q - rowi * 280) * 16;
        *(uint4*)(smem + rowi * LDS_ROW + off16) =
            *(const uint4*)(src + (size_t)rowi * (W_IN * C_IN) + off16);
      }
    }
  }
  __syncthreads();

  const int lane  = tid & 31;
  const int wv    = tid >> 5;
  const int mblk  = wv & 3;
  const int nhalf = wv >> 2;
  const int mrow  = lane & 15;
  const int khalf = lane >> 4;
  const int mcol0 = mblk * 16 + mrow;

  const uint4* wbase = wpack + (size_t)(nhalf * 4 * 32 + lane) * 2;

  v8i acc0 = {}; v8i acc1 = {}; v8i acc2 = {}; v8i acc3 = {};

#define LOAD_A(t, A)                                                          \
  {                                                                           \
    const int r_ = (t) / 7, s_ = (t) % 7;                                     \
    const unsigned char* ap_ =                                                \
        smem + r_ * LDS_ROW + (mcol0 + s_) * 64 + khalf * 8;                  \
    const uint2 a01_ = *(const uint2*)(ap_);                                  \
    const uint2 a23_ = *(const uint2*)(ap_ + 16);                             \
    const uint2 a45_ = *(const uint2*)(ap_ + 32);                             \
    const uint2 a67_ = *(const uint2*)(ap_ + 48);                             \
    A[0] = (int)a01_.x; A[1] = (int)a01_.y;                                   \
    A[2] = (int)a23_.x; A[3] = (int)a23_.y;                                   \
    A[4] = (int)a45_.x; A[5] = (int)a45_.y;                                   \
    A[6] = (int)a67_.x; A[7] = (int)a67_.y;                                   \
  }

#define LOAD_B(t, nb, B)                                                      \
  {                                                                           \
    const uint4* bp_ = wbase + (size_t)(t) * 512 + (nb) * 64;                 \
    const uint4 b0_ = bp_[0];                                                 \
    const uint4 b1_ = bp_[1];                                                 \
    B[0] = (int)b0_.x; B[1] = (int)b0_.y; B[2] = (int)b0_.z; B[3] = (int)b0_.w; \
    B[4] = (int)b1_.x; B[5] = (int)b1_.y; B[6] = (int)b1_.z; B[7] = (int)b1_.w; \
  }

#define LOAD_TAP(t, A, Bv0, Bv1, Bv2, Bv3)                                    \
  LOAD_A(t, A)                                                                \
  LOAD_B(t, 0, Bv0) LOAD_B(t, 1, Bv1) LOAD_B(t, 2, Bv2) LOAD_B(t, 3, Bv3)

#define WMMA4(A, Bv0, Bv1, Bv2, Bv3)                                          \
  acc0 = __builtin_amdgcn_wmma_i32_16x16x64_iu8(false, A, false, Bv0, acc0,   \
                                                false, false);                \
  acc1 = __builtin_amdgcn_wmma_i32_16x16x64_iu8(false, A, false, Bv1, acc1,   \
                                                false, false);                \
  acc2 = __builtin_amdgcn_wmma_i32_16x16x64_iu8(false, A, false, Bv2, acc2,   \
                                                false, false);                \
  acc3 = __builtin_amdgcn_wmma_i32_16x16x64_iu8(false, A, false, Bv3, acc3,   \
                                                false, false);

  v8i A0, A1;
  v8i B00, B01, B02, B03;
  v8i B10, B11, B12, B13;

  LOAD_TAP(0, A0, B00, B01, B02, B03)

#pragma unroll
  for (int t = 0; t < TAPS; t += 2) {
    if (t + 1 < TAPS) {
      LOAD_TAP(t + 1, A1, B10, B11, B12, B13)
    }
    WMMA4(A0, B00, B01, B02, B03)
    if (t + 1 < TAPS) {
      if (t + 2 < TAPS) {
        LOAD_TAP(t + 2, A0, B00, B01, B02, B03)
      }
      WMMA4(A1, B10, B11, B12, B13)
    }
  }

#undef LOAD_A
#undef LOAD_B
#undef LOAD_TAP
#undef WMMA4

  const int kb   = nhalf * 64 + mrow;
  const int jrow = j0 + mblk * 16 + khalf * 8;

#define STORE_NB(nb, ACC)                                                      \
  {                                                                            \
    const int k = kb + (nb) * 16;                                              \
    float* op = out + (((size_t)(n * K_OUT + k)) * H_OUT + io) * W_OUT + jrow; \
    op[0] = (float)ACC[0]; op[1] = (float)ACC[1];                              \
    op[2] = (float)ACC[2]; op[3] = (float)ACC[3];                              \
    op[4] = (float)ACC[4]; op[5] = (float)ACC[5];                              \
    op[6] = (float)ACC[6]; op[7] = (float)ACC[7];                              \
  }
  STORE_NB(0, acc0)
  STORE_NB(1, acc1)
  STORE_NB(2, acc2)
  STORE_NB(3, acc3)
#undef STORE_NB
}

extern "C" void kernel_launch(void* const* d_in, const int* in_sizes, int n_in,
                              void* d_out, int out_size, void* d_ws, size_t ws_size,
                              hipStream_t stream) {
  (void)in_sizes; (void)n_in; (void)out_size; (void)ws_size;
  const int*   x  = (const int*)d_in[0];
  const float* w1 = (const float*)d_in[1];
  const float* w2 = (const float*)d_in[2];
  const float* w3 = (const float*)d_in[3];
  float* out = (float*)d_out;

  unsigned char* xu8 = (unsigned char*)d_ws;
  unsigned int* wpack =
      (unsigned int*)((char*)d_ws + (size_t)N_IMG * H_IN * W_IN * C_IN);

  pack_x_kernel<<<N_IMG * H_IN, 128, 0, stream>>>(x, xu8);
  pack_w_kernel<<<TAPS, 256, 0, stream>>>(w1, w2, w3, wpack);
  conv7x7_iu8_wmma<<<dim3(2, H_OUT, N_IMG), 256, 0, stream>>>(
      xu8, (const uint4*)wpack, out);
}
